// DeployModel_38268158608230
// MI455X (gfx1250) — matrix operand order repaired and checked by reading, not compiled, not run on hardware
//
#include <hip/hip_runtime.h>
#include <hip/hip_bf16.h>
#include <math.h>

#define REG_MAX     16
#define NUM_CLASSES 80
#define IMG         640
#define PRE_TOP_K   1000
#define KEEP_TOP_K  100
#define IOU_THR     0.65f
#define SCORE_THR   0.25f
#define BATCH       16
#define N_ANCH      8400   // 80*80 + 40*40 + 20*20

typedef __attribute__((ext_vector_type(16))) _Float16 v16h;
typedef __attribute__((ext_vector_type(8)))  float    v8f;

// ---------------------------------------------------------------------------
// Decode one FPN level: DFL softmax-projection via V_WMMA_F32_16X16X32_F16,
// plus class max/argmax + sigmoid. One block = 256 threads = 8 waves,
// each wave handles 8 anchors (2 WMMA tiles of 4 anchors x 4 sides = 16 rows).
// H/W/stride are compile-time so index math constant-folds (no int division).
// ---------------------------------------------------------------------------
template <int H, int W, int STRIDE_I>
__global__ void __launch_bounds__(256)
decode_level_kernel(const float* __restrict__ cls,
                    const float* __restrict__ bbox,
                    int lvlOff,
                    float* __restrict__ boxes,   // [B*N_ANCH, 4]
                    float* __restrict__ scores,  // [B*N_ANCH]
                    int*   __restrict__ labels)  // [B*N_ANCH]
{
    constexpr int   HW     = H * W;
    constexpr float stride = (float)STRIDE_I;

    const int b    = blockIdx.y;
    const int lane = threadIdx.x & 31;
    const int wid  = threadIdx.x >> 5;
    const int tileBase = blockIdx.x * 64 + wid * 8;   // first anchor of this wave

    __shared__ float dred[8][256];   // per-wave WMMA D spill (32 lanes x 8 vgprs)
    __shared__ float sdist[8][16];   // per-wave decoded distances (4 anchors x 4 sides)

    const float* bb = bbox + (size_t)b * (4 * REG_MAX) * HW;
    const float* cl = cls  + (size_t)b * NUM_CLASSES * HW;

    // Prefetch class logits for this tile (gfx1250 global_prefetch_b8).
    {
        int ia0 = tileBase < HW ? tileBase : (HW - 1);
        __builtin_prefetch(cl + ia0, 0, 1);
    }

    // ---- B fragment: 32x16 f16. Column 0 = DFL projection (0..15),
    //      column 1 = ones; K = 8*(lane/16)+j in halves j=0..7, halves 8..15 (K=16..31) zero.
    v16h bfrag = {};
    if ((lane & 15) == 0) {
        #pragma unroll
        for (int k = 0; k < 8; ++k) bfrag[k] = (_Float16)(float)(k + (lane >> 4) * 8);
    } else if ((lane & 15) == 1) {
        #pragma unroll
        for (int k = 0; k < 8; ++k) bfrag[k] = (_Float16)1.0f;
    }

    // ---- DFL decode: two WMMA tiles of 4 anchors each ----
    #pragma unroll
    for (int sub = 0; sub < 2; ++sub) {
        // A-matrix row assignment (16-bit A layout, 16x32):
        //   lanes 0-15  : row m = lane,    K = 0..7  in halves 0..7
        //   lanes 16-31 : row m = lane-16, K = 8..15 in halves 0..7
        //   halves 8..15 (K=16..31) = 0
        const int m  = lane & 15;        // row = anchorLocal*4 + side
        const int aL = m >> 2;
        const int s  = m & 3;
        const int kb = (lane >> 4) * 8;
        int ia  = tileBase + sub * 4 + aL;
        int iac = (ia < HW) ? ia : (HW - 1);

        float x[8];
        const float* p = bb + (size_t)(s * REG_MAX + kb) * HW + iac;
        #pragma unroll
        for (int j = 0; j < 8; ++j) x[j] = p[(size_t)j * HW];

        // row max for numerically stable softmax (row split across lane pair L, L^16)
        float mloc = x[0];
        #pragma unroll
        for (int j = 1; j < 8; ++j) mloc = fmaxf(mloc, x[j]);
        float mrow = fmaxf(mloc, __shfl_xor(mloc, 16, 32));

        v16h afrag = {};
        #pragma unroll
        for (int j = 0; j < 8; ++j) afrag[j] = (_Float16)__expf(x[j] - mrow);

        v8f c = {};
        // D = A(16x32 exp) * B(32x16 [proj|ones|0...]) : col0 = sum k*e, col1 = sum e
        c = __builtin_amdgcn_wmma_f32_16x16x32_f16(
                /*neg_a=*/false, afrag, /*neg_b=*/false, bfrag,
                /*c_mod=*/(short)0, c, /*reuse_a=*/false, /*reuse_b=*/false);

        #pragma unroll
        for (int r = 0; r < 8; ++r) dred[wid][lane * 8 + r] = c[r];

        // Extract D(m,0)/D(m,1): element (M,N) lives in lane N+(M<8?0:16), vgpr M&7.
        if (lane < 16) {
            int srcLane = (m < 8) ? 0 : 16;
            float num = dred[wid][(srcLane + 0) * 8 + (m & 7)];
            float den = dred[wid][(srcLane + 1) * 8 + (m & 7)];
            // fast reciprocal (v_rcp_f32) instead of IEEE divide sequence
            sdist[wid][m] = num * __builtin_amdgcn_rcpf(den) * stride;
        }

        if (lane < 4) {
            int ia2 = tileBase + sub * 4 + lane;
            if (ia2 < HW) {
                int hh = ia2 / W, ww = ia2 % W;   // W constexpr -> magic multiply
                float px = (ww + 0.5f) * stride;
                float py = (hh + 0.5f) * stride;
                float d0 = sdist[wid][lane * 4 + 0];
                float d1 = sdist[wid][lane * 4 + 1];
                float d2 = sdist[wid][lane * 4 + 2];
                float d3 = sdist[wid][lane * 4 + 3];
                int gi = b * N_ANCH + lvlOff + ia2;
                float4 bx = make_float4(px - d0, py - d1, px + d2, py + d3);
                ((float4*)boxes)[gi] = bx;
            }
        }
    }

    // ---- Class scores: 4 lanes per anchor scan 80 channels ----
    {
        const int g  = lane >> 2;           // anchor 0..7 within wave tile
        const int q  = lane & 3;
        int ia  = tileBase + g;
        int iac = (ia < HW) ? ia : (HW - 1);

        float best = -3.4e38f;
        int   bidx = 0;
        #pragma unroll 4
        for (int cch = q; cch < NUM_CLASSES; cch += 4) {
            float v = cl[(size_t)cch * HW + iac];
            if (v > best) { best = v; bidx = cch; }
        }
        #pragma unroll
        for (int off = 1; off < 4; off <<= 1) {
            float ov = __shfl_xor(best, off, 32);
            int   oi = __shfl_xor(bidx, off, 32);
            if (ov > best || (ov == best && oi < bidx)) { best = ov; bidx = oi; }
        }
        if (q == 0 && ia < HW) {
            int gi = b * N_ANCH + lvlOff + ia;
            // sigmoid(max logit) with fast rcp
            scores[gi] = __builtin_amdgcn_rcpf(1.0f + __expf(-best));
            labels[gi] = bidx;
        }
    }

    __builtin_amdgcn_s_wait_tensorcnt(0);
}

// ---------------------------------------------------------------------------
__global__ void init_kernel(int* counters) {
    if (threadIdx.x < BATCH) counters[threadIdx.x] = 0;
}

// Threshold + compaction of candidates per image (cap PRE_TOP_K).
__global__ void __launch_bounds__(256)
compact_kernel(const float* __restrict__ scores,
               const int*   __restrict__ labels,
               const float* __restrict__ boxes,
               float* __restrict__ cScores,
               int*   __restrict__ cLabels,
               float* __restrict__ cBoxes,
               int*   __restrict__ counters)
{
    const int b = blockIdx.x;
    for (int i = threadIdx.x; i < N_ANCH; i += blockDim.x) {
        float s = scores[b * N_ANCH + i];
        if (s > SCORE_THR) {
            int slot = atomicAdd(&counters[b], 1);
            if (slot < PRE_TOP_K) {
                cScores[b * PRE_TOP_K + slot] = s;
                cLabels[b * PRE_TOP_K + slot] = labels[b * N_ANCH + i];
                const float4 bx = ((const float4*)boxes)[b * N_ANCH + i];
                ((float4*)cBoxes)[b * PRE_TOP_K + slot] = bx;
            }
        }
    }
}

// Greedy class-aware NMS, one block per image, KEEP_TOP_K iterations.
__global__ void __launch_bounds__(256)
nms_kernel(float* __restrict__ cScores,
           const int*   __restrict__ cLabels,
           const float* __restrict__ cBoxes,
           const int*   __restrict__ counters,
           float* __restrict__ outBoxes,    // [B,100,4]
           float* __restrict__ outScores,   // [B,100]
           float* __restrict__ outLabels)   // [B,100]
{
    __shared__ float sval[256];
    __shared__ int   sidx[256];
    __shared__ float bb4[4];
    __shared__ int   act;

    const int b   = blockIdx.x;
    const int tid = threadIdx.x;
    int cnt = counters[b];
    if (cnt > PRE_TOP_K) cnt = PRE_TOP_K;

    float*       S  = cScores + b * PRE_TOP_K;
    const float* BX = cBoxes  + (size_t)b * PRE_TOP_K * 4;
    const int*   L  = cLabels + b * PRE_TOP_K;

    for (int it = 0; it < KEEP_TOP_K; ++it) {
        // parallel argmax over candidates
        float best = -3.4e38f; int bidx = 0;
        for (int e = tid; e < cnt; e += 256) {
            float v = S[e];
            if (v > best) { best = v; bidx = e; }
        }
        sval[tid] = best; sidx[tid] = bidx;
        __syncthreads();
        for (int off = 128; off > 0; off >>= 1) {
            if (tid < off) {
                if (sval[tid + off] > sval[tid]) {
                    sval[tid] = sval[tid + off];
                    sidx[tid] = sidx[tid + off];
                }
            }
            __syncthreads();
        }

        if (tid == 0) {
            int idx = sidx[0];
            float v = (cnt > 0) ? sval[0] : 0.0f;
            float x1 = 0.f, y1 = 0.f, x2 = 0.f, y2 = 0.f;
            int lab = 0;
            if (cnt > 0) {
                x1 = BX[idx * 4 + 0]; y1 = BX[idx * 4 + 1];
                x2 = BX[idx * 4 + 2]; y2 = BX[idx * 4 + 3];
                lab = L[idx];
            }
            float* ob = outBoxes + (size_t)(b * KEEP_TOP_K + it) * 4;
            ob[0] = x1; ob[1] = y1; ob[2] = x2; ob[3] = y2;
            outScores[b * KEEP_TOP_K + it] = (v > 0.f) ? v : 0.f;
            outLabels[b * KEEP_TOP_K + it] = (float)lab;
            float co = (float)lab * (IMG * 8.0f);   // class-offset trick
            bb4[0] = x1 + co; bb4[1] = y1 + co; bb4[2] = x2 + co; bb4[3] = y2 + co;
            act = (cnt > 0 && v > 0.f) ? 1 : 0;
        }
        __syncthreads();

        if (act) {
            float ax1 = bb4[0], ay1 = bb4[1], ax2 = bb4[2], ay2 = bb4[3];
            float aarea = (ax2 - ax1) * (ay2 - ay1);
            for (int e = tid; e < cnt; e += 256) {
                float co = (float)L[e] * (IMG * 8.0f);
                float x1 = BX[e * 4 + 0] + co, y1 = BX[e * 4 + 1] + co;
                float x2 = BX[e * 4 + 2] + co, y2 = BX[e * 4 + 3] + co;
                float lt0 = fmaxf(ax1, x1), lt1 = fmaxf(ay1, y1);
                float rb0 = fminf(ax2, x2), rb1 = fminf(ay2, y2);
                float w = fmaxf(rb0 - lt0, 0.f), h = fmaxf(rb1 - lt1, 0.f);
                float inter = w * h;
                float area  = (x2 - x1) * (y2 - y1);
                float iou   = inter * __builtin_amdgcn_rcpf(aarea + area - inter + 1e-7f);
                if (iou > IOU_THR) S[e] = -1.0f;   // self has iou 1 -> suppressed
            }
        }
        __syncthreads();
    }
}

// ---------------------------------------------------------------------------
extern "C" void kernel_launch(void* const* d_in, const int* in_sizes, int n_in,
                              void* d_out, int out_size, void* d_ws, size_t ws_size,
                              hipStream_t stream)
{
    const float* cls8  = (const float*)d_in[0];
    const float* bb8   = (const float*)d_in[1];
    const float* cls16 = (const float*)d_in[2];
    const float* bb16  = (const float*)d_in[3];
    const float* cls32 = (const float*)d_in[4];
    const float* bb32  = (const float*)d_in[5];
    float* out = (float*)d_out;

    char* ws = (char*)d_ws;
    size_t off = 0;
    float* boxes   = (float*)(ws + off); off += (size_t)BATCH * N_ANCH * 4 * sizeof(float);
    float* scores  = (float*)(ws + off); off += (size_t)BATCH * N_ANCH * sizeof(float);
    int*   labels  = (int*)  (ws + off); off += (size_t)BATCH * N_ANCH * sizeof(int);
    float* cBoxes  = (float*)(ws + off); off += (size_t)BATCH * PRE_TOP_K * 4 * sizeof(float);
    float* cScores = (float*)(ws + off); off += (size_t)BATCH * PRE_TOP_K * sizeof(float);
    int*   cLabels = (int*)  (ws + off); off += (size_t)BATCH * PRE_TOP_K * sizeof(int);
    int*   counters= (int*)  (ws + off); off += (size_t)BATCH * sizeof(int);
    (void)ws_size; (void)in_sizes; (void)n_in; (void)out_size;

    init_kernel<<<1, 32, 0, stream>>>(counters);

    // s8: 80x80=6400 anchors -> 100 blocks; s16: 1600 -> 25; s32: 400 -> 7 (clamped)
    decode_level_kernel<80, 80, 8><<<dim3(100, BATCH), 256, 0, stream>>>(
        cls8,  bb8,  0,    boxes, scores, labels);
    decode_level_kernel<40, 40, 16><<<dim3(25, BATCH), 256, 0, stream>>>(
        cls16, bb16, 6400, boxes, scores, labels);
    decode_level_kernel<20, 20, 32><<<dim3(7, BATCH), 256, 0, stream>>>(
        cls32, bb32, 8000, boxes, scores, labels);

    compact_kernel<<<BATCH, 256, 0, stream>>>(scores, labels, boxes,
                                              cScores, cLabels, cBoxes, counters);

    float* outBoxes  = out;
    float* outScores = out + (size_t)BATCH * KEEP_TOP_K * 4;
    float* outLabels = out + (size_t)BATCH * KEEP_TOP_K * 5;
    nms_kernel<<<BATCH, 256, 0, stream>>>(cScores, cLabels, cBoxes, counters,
                                          outBoxes, outScores, outLabels);
}
